// MoESystem_62474594287692
// MI455X (gfx1250) — matrix operand order repaired and checked by reading, not compiled, not run on hardware
//
#include <hip/hip_runtime.h>
#include <hip/hip_bf16.h>

typedef __attribute__((ext_vector_type(16))) __bf16 v16bf;
typedef __attribute__((ext_vector_type(8)))  float  v8f;

#define TT 48
#define DD 32
#define KK 1536      // T*D
#define NN 1024      // hidden H
#define GH 128       // gate hidden
#define BM 128
#define BN 128
#define BK 64        // staged K depth = 2 WMMA k-steps

__device__ __forceinline__ unsigned short f2bf(float f) {
    unsigned int u = __float_as_uint(f);
    unsigned int r = (u + 0x7FFFu + ((u >> 16) & 1u)) >> 16;
    return (unsigned short)r;
}

// gfx1250 async global->LDS copy (bypasses VGPRs, tracked by ASYNCcnt).
// Low 32 bits of a generic shared-memory pointer are the LDS byte offset
// (ISA 10.2).
__device__ __forceinline__ void async_cp16(void* lds, const void* g) {
    unsigned loff = (unsigned)(uintptr_t)lds;
    asm volatile("global_load_async_to_lds_b128 %0, %1, off"
                 :: "v"(loff), "v"(g) : "memory");
}

// ---- f32 -> bf16 bulk convert (x_flat), 4 elements/thread ----
__global__ void cvt_x_kernel(const float* __restrict__ x,
                             unsigned short* __restrict__ xb, int n) {
    int i = (blockIdx.x * blockDim.x + threadIdx.x) * 4;
    if (i + 3 < n) {
        float4 v = *(const float4*)(x + i);
        ushort4 o;
        o.x = f2bf(v.x); o.y = f2bf(v.y); o.z = f2bf(v.z); o.w = f2bf(v.w);
        *(ushort4*)(xb + i) = o;
    }
}

// ---- W1 [K][N] f32 -> W1T [N][K] bf16 (transpose so B-frags are contiguous in K) ----
__global__ void cvt_wT_kernel(const float* __restrict__ w,
                              unsigned short* __restrict__ wt, int K, int N) {
    int idx = blockIdx.x * blockDim.x + threadIdx.x;
    if (idx < K * N) {
        int k = idx / N, n = idx % N;
        wt[(size_t)n * K + k] = f2bf(w[idx]);
    }
}

// ---- gate: one wave per sample ----
__global__ __launch_bounds__(256) void gate_kernel(
    const float* __restrict__ x,
    const float* __restrict__ gW1, const float* __restrict__ gb1,
    const float* __restrict__ gW2, const float* __restrict__ gb2,
    float* __restrict__ probs, int* __restrict__ decs, int B) {
    __shared__ float m[8][DD];
    const int wave = threadIdx.x >> 5, lane = threadIdx.x & 31;
    const int b = blockIdx.x * 8 + wave;
    if (b >= B) return;

    // mean-pool over T: lane = channel d, coalesced across lanes
    const float* xb = x + (size_t)b * KK;
    float s = 0.f;
    #pragma unroll
    for (int t = 0; t < TT; ++t) s += xb[t * DD + lane];
    m[wave][lane] = s * (1.0f / (float)TT);
    __syncthreads();

    // hidden 32->128 (4 units per lane) + logit partials
    float z0 = 0.f, z1 = 0.f;
    #pragma unroll
    for (int i = 0; i < 4; ++i) {
        int j = lane + i * 32;
        float h = gb1[j];
        #pragma unroll
        for (int d = 0; d < DD; ++d) h += m[wave][d] * gW1[d * GH + j];
        h = fmaxf(h, 0.f);
        z0 += h * gW2[j * 2 + 0];
        z1 += h * gW2[j * 2 + 1];
    }
    #pragma unroll
    for (int off = 16; off > 0; off >>= 1) {
        z0 += __shfl_xor(z0, off, 32);
        z1 += __shfl_xor(z1, off, 32);
    }
    if (lane == 0) {
        z0 += gb2[0]; z1 += gb2[1];
        float mx = fmaxf(z0, z1);
        float e0 = __expf(z0 - mx), e1 = __expf(z1 - mx);
        probs[b] = fmaxf(e0, e1) / (e0 + e1);
        decs[b]  = (z1 > z0) ? 1 : 0;
    }
}

// ---- fused expert: relu(Xbf16 @ W1T + b1) @ W2 + b2, layer2 folded into GEMM ----
// Double-buffered LDS, async global->LDS staging, bf16 WMMA, f32 accum.
__global__ __launch_bounds__(256, 1) void expert_kernel(
    const unsigned short* __restrict__ xb,   // [B][1536] bf16
    const unsigned short* __restrict__ w1t,  // [2][1024][1536] bf16
    const float* __restrict__ b1n, const float* __restrict__ b1a,
    const float* __restrict__ w2n, const float* __restrict__ w2a,
    const float* __restrict__ b2n, const float* __restrict__ b2a,
    float* __restrict__ eout,                // [2][B][2]
    int Btot) {
    const int expert = blockIdx.y;
    const unsigned short* wt = w1t + (size_t)expert * NN * KK;
    const float* b1 = expert ? b1a : b1n;
    const float* w2 = expert ? w2a : w2n;
    const float* b2 = expert ? b2a : b2n;
    const int row0 = blockIdx.x * BM;

    __shared__ __align__(16) unsigned short lA[2][BM * BK]; // [row][k]
    __shared__ __align__(16) unsigned short lB[2][BN * BK]; // [n][k] (W1 transposed)
    __shared__ float lOut[BM * 2];

    const int tid = threadIdx.x;
    const int lane = tid & 31, wave = tid >> 5;
    const int wr = (wave & 3) * 32;   // wave's row offset in 128-row tile
    const int wc = (wave >> 2) * 64;  // wave's col offset in 128-col n-block
    const int hi = lane >> 4, lidx = lane & 15;

    if (tid < BM * 2) lOut[tid] = 0.f;

    // per-lane layer-2 partials: (row-tile a, acc element i, class k2)
    float pacc[2][8][2];
    #pragma unroll
    for (int a = 0; a < 2; ++a)
        #pragma unroll
        for (int i = 0; i < 8; ++i) { pacc[a][i][0] = 0.f; pacc[a][i][1] = 0.f; }

    const int srow = tid >> 1, half = tid & 1;   // staging: 2 threads/row, 32 elems each
    const int NITER = KK / BK;                   // 24

    for (int nblk = 0; nblk < NN; nblk += BN) {
        v8f acc[2][4];
        #pragma unroll
        for (int a = 0; a < 2; ++a)
            #pragma unroll
            for (int c = 0; c < 4; ++c) { v8f z = {}; acc[a][c] = z; }

        // prologue: async-stage tile 0 into buffer 0 (8x b128 per thread)
        {
            const unsigned short* gA = xb + (size_t)(row0 + srow) * KK + half * 32;
            const unsigned short* gB = wt + (size_t)(nblk + srow) * KK + half * 32;
            #pragma unroll
            for (int c = 0; c < 4; ++c) {
                async_cp16(&lA[0][srow * BK + half * 32 + c * 8], gA + c * 8);
                async_cp16(&lB[0][srow * BK + half * 32 + c * 8], gB + c * 8);
            }
        }

        for (int ki = 0; ki < NITER; ++ki) {
            const int cur = ki & 1;
            if (ki + 1 < NITER) {
                // issue next tile into the other buffer, then wait for current
                const int kn = (ki + 1) * BK;
                const unsigned short* gA = xb + (size_t)(row0 + srow) * KK + kn + half * 32;
                const unsigned short* gB = wt + (size_t)(nblk + srow) * KK + kn + half * 32;
                #pragma unroll
                for (int c = 0; c < 4; ++c) {
                    async_cp16(&lA[cur ^ 1][srow * BK + half * 32 + c * 8], gA + c * 8);
                    async_cp16(&lB[cur ^ 1][srow * BK + half * 32 + c * 8], gB + c * 8);
                }
                asm volatile("s_wait_asynccnt 0x8" ::: "memory");
            } else {
                asm volatile("s_wait_asynccnt 0x0" ::: "memory");
            }
            __syncthreads();

            union Frag { v16bf v; uint4 q[2]; };
            #pragma unroll
            for (int k0 = 0; k0 < BK; k0 += 32) {
                Frag afrag[2], bfrag[4];
                // A (16-bit 16x32): lane<16 -> K {0..7,16..23}, lane>=16 -> K {8..15,24..31}
                #pragma unroll
                for (int a = 0; a < 2; ++a) {
                    int row = wr + a * 16 + lidx;
                    afrag[a].q[0] = *(const uint4*)(&lA[cur][row * BK + k0 + hi * 8]);
                    afrag[a].q[1] = *(const uint4*)(&lA[cur][row * BK + k0 + 16 + hi * 8]);
                }
                // B (32x16): col = lane&15; lane<16 -> K {0..7,16..23}, lane>=16 -> K {8..15,24..31}
                #pragma unroll
                for (int c = 0; c < 4; ++c) {
                    int n = wc + c * 16 + lidx;
                    bfrag[c].q[0] = *(const uint4*)(&lB[cur][n * BK + k0 + hi * 8]);
                    bfrag[c].q[1] = *(const uint4*)(&lB[cur][n * BK + k0 + 16 + hi * 8]);
                }
                #pragma unroll
                for (int a = 0; a < 2; ++a)
                    #pragma unroll
                    for (int c = 0; c < 4; ++c)
                        acc[a][c] = __builtin_amdgcn_wmma_f32_16x16x32_bf16(
                            false, afrag[a].v, false, bfrag[c].v,
                            (short)0, acc[a][c], false, false);
            }
            __syncthreads();
        }

        // fold layer 2: h = relu(acc + b1[col]); pacc += h * W2[col][:]
        #pragma unroll
        for (int c = 0; c < 4; ++c) {
            int col = nblk + wc + c * 16 + lidx;
            float bb  = b1[col];
            float w20 = w2[col * 2 + 0];
            float w21 = w2[col * 2 + 1];
            #pragma unroll
            for (int a = 0; a < 2; ++a)
                #pragma unroll
                for (int i = 0; i < 8; ++i) {
                    float h = fmaxf(acc[a][c][i] + bb, 0.f);
                    pacc[a][i][0] += h * w20;
                    pacc[a][i][1] += h * w21;
                }
        }
    }

    // cross-lane / cross-wave reduction through LDS atomics
    #pragma unroll
    for (int a = 0; a < 2; ++a)
        #pragma unroll
        for (int i = 0; i < 8; ++i) {
            int row = wr + a * 16 + hi * 8 + i;  // C layout: M = i + 8*hi
            atomicAdd(&lOut[row * 2 + 0], pacc[a][i][0]);
            atomicAdd(&lOut[row * 2 + 1], pacc[a][i][1]);
        }
    __syncthreads();

    if (tid < BM * 2) {
        int row = tid >> 1, k2 = tid & 1;
        eout[((size_t)expert * Btot + row0 + row) * 2 + k2] = lOut[tid] + b2[k2];
    }
}

// ---- routing combine ----
__global__ void combine_kernel(const float* __restrict__ eout,
                               const float* __restrict__ probs,
                               const int* __restrict__ decs,
                               float* __restrict__ out, int B) {
    int b = blockIdx.x * blockDim.x + threadIdx.x;
    if (b >= B) return;
    bool abn = (probs[b] >= 0.7f) && (decs[b] == 1);
    const float* src = eout + (abn ? (size_t)B * 2 : 0);
    out[b * 2 + 0] = src[b * 2 + 0];
    out[b * 2 + 1] = src[b * 2 + 1];
}

extern "C" void kernel_launch(void* const* d_in, const int* in_sizes, int n_in,
                              void* d_out, int out_size, void* d_ws, size_t ws_size,
                              hipStream_t stream) {
    const float* x   = (const float*)d_in[0];
    const float* gW1 = (const float*)d_in[1];
    const float* gb1 = (const float*)d_in[2];
    const float* gW2 = (const float*)d_in[3];
    const float* gb2 = (const float*)d_in[4];
    const float* nW1 = (const float*)d_in[5];
    const float* nb1 = (const float*)d_in[6];
    const float* nW2 = (const float*)d_in[7];
    const float* nb2 = (const float*)d_in[8];
    const float* aW1 = (const float*)d_in[9];
    const float* ab1 = (const float*)d_in[10];
    const float* aW2 = (const float*)d_in[11];
    const float* ab2 = (const float*)d_in[12];

    const int B = in_sizes[0] / KK;   // 16384

    // workspace layout
    char* ws = (char*)d_ws;
    unsigned short* xb  = (unsigned short*)ws;                     // B*1536 bf16
    size_t off = (size_t)B * KK * sizeof(unsigned short);
    unsigned short* w1t = (unsigned short*)(ws + off);             // 2*1024*1536 bf16
    off += (size_t)2 * NN * KK * sizeof(unsigned short);
    float* eout = (float*)(ws + off);                              // 2*B*2 f32

    // d_out layout: [outputs B*2][decisions B][probs B]
    float* out   = (float*)d_out;
    int*   decs  = (int*)((float*)d_out + (size_t)2 * B);
    float* probs = (float*)d_out + (size_t)3 * B;

    // 1) precision conversion
    {
        int n = B * KK;
        cvt_x_kernel<<<(n / 4 + 255) / 256, 256, 0, stream>>>(x, xb, n);
        int m = KK * NN;
        cvt_wT_kernel<<<(m + 255) / 256, 256, 0, stream>>>(nW1, w1t, KK, NN);
        cvt_wT_kernel<<<(m + 255) / 256, 256, 0, stream>>>(aW1, w1t + (size_t)NN * KK, KK, NN);
    }
    // 2) gate
    gate_kernel<<<(B + 7) / 8, 256, 0, stream>>>(x, gW1, gb1, gW2, gb2, probs, decs, B);
    // 3) fused experts (WMMA + async LDS staging)
    {
        dim3 grid(B / BM, 2);
        expert_kernel<<<grid, 256, 0, stream>>>(xb, w1t, nb1, ab1, nW2, aW2, nb2, ab2, eout, B);
    }
    // 4) routing combine
    combine_kernel<<<(B + 255) / 256, 256, 0, stream>>>(eout, probs, decs, out, B);
}
